// GraphEncoder_19731079757865
// MI455X (gfx1250) — matrix operand order repaired and checked by reading, not compiled, not run on hardware
//
#include <hip/hip_runtime.h>
#include <hip/hip_bf16.h>

typedef __bf16 bf16_t;
typedef __attribute__((ext_vector_type(4)))  bf16_t v4bf;
typedef __attribute__((ext_vector_type(8)))  bf16_t v8bf;
typedef __attribute__((ext_vector_type(16))) bf16_t v16bf;
typedef __attribute__((ext_vector_type(8)))  float  v8f;

#define IN_DIM  1024
#define HID_DIM 256
#define OUT_DIM 512

// ---------------- utility kernels ----------------
__global__ void k_zero_f32(float* __restrict__ p, long n) {
  long i  = (long)blockIdx.x * blockDim.x + threadIdx.x;
  long st = (long)gridDim.x * blockDim.x;
  for (; i < n; i += st) p[i] = 0.0f;
}

__global__ void k_cvt_bf16(const float* __restrict__ s, bf16_t* __restrict__ d, long n8) {
  long t = (long)blockIdx.x * blockDim.x + threadIdx.x;
  if (t >= n8) return;
  long i = t * 8;
  float4 a = *(const float4*)(s + i);
  float4 b = *(const float4*)(s + i + 4);
  v8bf o;
  o[0]=(bf16_t)a.x; o[1]=(bf16_t)a.y; o[2]=(bf16_t)a.z; o[3]=(bf16_t)a.w;
  o[4]=(bf16_t)b.x; o[5]=(bf16_t)b.y; o[6]=(bf16_t)b.z; o[7]=(bf16_t)b.w;
  *(v8bf*)(d + i) = o;
}

// d has 2*rows_each rows of rowlen: first rows_each rows from s0, rest from s1 (stack along N)
__global__ void k_pack_rows_bf16(const float* __restrict__ s0, const float* __restrict__ s1,
                                 int rows_each, int rowlen, bf16_t* __restrict__ d) {
  long t  = (long)blockIdx.x * blockDim.x + threadIdx.x;
  long n8 = (long)2 * rows_each * rowlen / 8;
  if (t >= n8) return;
  long i = t * 8;
  long r = i / rowlen;
  long k = i % rowlen;
  const float* src = (r < rows_each) ? (s0 + r * (long)rowlen + k)
                                     : (s1 + (r - rows_each) * (long)rowlen + k);
  float4 a = *(const float4*)(src);
  float4 b = *(const float4*)(src + 4);
  v8bf o;
  o[0]=(bf16_t)a.x; o[1]=(bf16_t)a.y; o[2]=(bf16_t)a.z; o[3]=(bf16_t)a.w;
  o[4]=(bf16_t)b.x; o[5]=(bf16_t)b.y; o[6]=(bf16_t)b.z; o[7]=(bf16_t)b.w;
  *(v8bf*)(d + i) = o;
}

// d[r][k] = k < half ? s0[r][k] : s1[r][k-half]  (concat along K), rowlen = 2*half
__global__ void k_pack_cols_bf16(const float* __restrict__ s0, const float* __restrict__ s1,
                                 int rows, int half, bf16_t* __restrict__ d) {
  long t      = (long)blockIdx.x * blockDim.x + threadIdx.x;
  int  rowlen = 2 * half;
  long n8     = (long)rows * rowlen / 8;
  if (t >= n8) return;
  long i = t * 8;
  long r = i / rowlen;
  int  k = (int)(i % rowlen);
  const float* src = (k < half) ? (s0 + r * (long)half + k)
                                : (s1 + r * (long)half + (k - half));
  float4 a = *(const float4*)(src);
  float4 b = *(const float4*)(src + 4);
  v8bf o;
  o[0]=(bf16_t)a.x; o[1]=(bf16_t)a.y; o[2]=(bf16_t)a.z; o[3]=(bf16_t)a.w;
  o[4]=(bf16_t)b.x; o[5]=(bf16_t)b.y; o[6]=(bf16_t)b.z; o[7]=(bf16_t)b.w;
  *(v8bf*)(d + i) = o;
}

// ---------------- bf16 WMMA GEMM: C[M,N] = A[M,K] @ B[N,K]^T (+bias) ----------------
// Block: 256 threads = 8 waves (wave32); wave grid 4(M) x 2(N); wave tile 32x64; block tile 128x128.
__global__ __launch_bounds__(256)
void k_gemm_bf16_wmma(const bf16_t* __restrict__ A, const bf16_t* __restrict__ B,
                      float* __restrict__ C, const float* __restrict__ bias,
                      int M, int N, int K) {
  const int lane = threadIdx.x & 31;
  const int wave = threadIdx.x >> 5;
  const int wm = wave & 3, wn = wave >> 2;
  const int m0 = blockIdx.y * 128 + wm * 32;
  const int n0 = blockIdx.x * 128 + wn * 64;

  v8f acc[2][4] = {};

  // A fragment addressing (16-bit A 16x32 layout): row = lane&15 (+16 per mi),
  // per-lane K chunks of 8 at (lane>>4)*8 and +16.
  const int akoff = (lane >> 4) * 8;
  const bf16_t* aptr[2];
#pragma unroll
  for (int mi = 0; mi < 2; ++mi) {
    int r = m0 + mi * 16 + (lane & 15);
    if (r > M - 1) r = M - 1;           // clamp tail rows; stores are guarded below
    aptr[mi] = A + (long)r * K + akoff;
  }
  // B fragment addressing (16-bit B 32x16 layout): col N = lane&15,
  // per-lane K chunks of 8 at (lane>>4)*8 and +16. B is [N,K] row-major.
  const int bkoff = (lane >> 4) * 8;
  const bf16_t* bptr[4];
#pragma unroll
  for (int ni = 0; ni < 4; ++ni)
    bptr[ni] = B + (long)(n0 + ni * 16 + (lane & 15)) * K + bkoff;

  for (int kk = 0; kk < K; kk += 32) {
    v16bf af[2], bfr[4];
#pragma unroll
    for (int mi = 0; mi < 2; ++mi) {
      v8bf lo = *(const v8bf*)(aptr[mi] + kk);
      v8bf hi = *(const v8bf*)(aptr[mi] + kk + 16);
      af[mi] = __builtin_shufflevector(lo, hi, 0,1,2,3,4,5,6,7,8,9,10,11,12,13,14,15);
    }
#pragma unroll
    for (int ni = 0; ni < 4; ++ni) {
      v8bf lo = *(const v8bf*)(bptr[ni] + kk);
      v8bf hi = *(const v8bf*)(bptr[ni] + kk + 16);
      bfr[ni] = __builtin_shufflevector(lo, hi, 0,1,2,3,4,5,6,7,8,9,10,11,12,13,14,15);
    }
    if (kk + 128 < K) {                  // global_prefetch_b8 four tiles ahead
      __builtin_prefetch(aptr[0] + kk + 128, 0, 0);
      __builtin_prefetch(aptr[1] + kk + 128, 0, 0);
    }
#pragma unroll
    for (int mi = 0; mi < 2; ++mi)
#pragma unroll
      for (int ni = 0; ni < 4; ++ni)
        acc[mi][ni] = __builtin_amdgcn_wmma_f32_16x16x32_bf16(
            false, af[mi], false, bfr[ni], (short)0, acc[mi][ni], false, false);
  }

  // C/D layout: VGPR r -> lanes 0-15: M=r, lanes 16-31: M=r+8; N = lane&15
  const int rbase = m0 + ((lane >> 4) << 3);
  const int col0  = n0 + (lane & 15);
#pragma unroll
  for (int mi = 0; mi < 2; ++mi) {
#pragma unroll
    for (int r = 0; r < 8; ++r) {
      int row = rbase + mi * 16 + r;
      if (row < M) {
#pragma unroll
        for (int ni = 0; ni < 4; ++ni) {
          int col = col0 + ni * 16;
          float v = acc[mi][ni][r];
          if (bias) v += bias[col];
          C[(long)row * N + col] = v;
        }
      }
    }
  }
}

// ---------------- graph aggregation kernels ----------------
__global__ void k_count_edges(const long long* __restrict__ dst, float* __restrict__ cnt, int E) {
  int e = blockIdx.x * blockDim.x + threadIdx.x;
  if (e < E) atomicAdd(&cnt[dst[e]], 1.0f);
}

// one wave per edge; gather 256 f32 from feat[src][coloff:coloff+256], atomic-add into agg[dst]
__global__ void k_scatter_f32(const long long* __restrict__ src, const long long* __restrict__ dst,
                              const float* __restrict__ feat, int ld, int coloff,
                              float* __restrict__ agg, int E) {
  int gwave  = (int)((blockIdx.x * blockDim.x + threadIdx.x) >> 5);
  int lane   = threadIdx.x & 31;
  int nwaves = (int)((gridDim.x * blockDim.x) >> 5);
  for (int e = gwave; e < E; e += nwaves) {
    long s = (long)src[e], d = (long)dst[e];
    const float4* frow = (const float4*)(feat + s * (long)ld + coloff);
    float* arow = agg + d * 256L;
#pragma unroll
    for (int i = 0; i < 2; ++i) {
      int c = lane + i * 32;          // 64 float4 = 256 floats
      float4 v = frow[c];
      atomicAdd(arow + 4 * c + 0, v.x);
      atomicAdd(arow + 4 * c + 1, v.y);
      atomicAdd(arow + 4 * c + 2, v.z);
      atomicAdd(arow + 4 * c + 3, v.w);
    }
  }
}

// one wave per edge; gather 256 bf16 from feat[src][coloff:], atomic-add f32 into agg[dst]
__global__ void k_scatter_bf16(const long long* __restrict__ src, const long long* __restrict__ dst,
                               const bf16_t* __restrict__ feat, int ld, int coloff,
                               float* __restrict__ agg, int E) {
  int gwave  = (int)((blockIdx.x * blockDim.x + threadIdx.x) >> 5);
  int lane   = threadIdx.x & 31;
  int nwaves = (int)((gridDim.x * blockDim.x) >> 5);
  for (int e = gwave; e < E; e += nwaves) {
    long s = (long)src[e], d = (long)dst[e];
    const v8bf v = *(const v8bf*)(feat + s * (long)ld + coloff + lane * 8);
    float* arow = agg + d * 256L + lane * 8;
#pragma unroll
    for (int j = 0; j < 8; ++j) atomicAdd(arow + j, (float)v[j]);
  }
}

// h = relu(agg/max(cnt,1) + y1[:,256:512] + b1) -> zb[:,256:512] (bf16)
__global__ void k_combine1(const float* __restrict__ y1, const float* __restrict__ agg,
                           const float* __restrict__ cnt, const float* __restrict__ b1,
                           bf16_t* __restrict__ zb, int NN) {
  long t = (long)blockIdx.x * blockDim.x + threadIdx.x;
  long total = (long)NN * 64;
  if (t >= total) return;
  int i  = (int)(t >> 6);
  int c4 = (int)(t & 63) * 4;
  float inv = 1.0f / fmaxf(cnt[i], 1.0f);
  float4 a = *(const float4*)(agg + (long)i * 256 + c4);
  float4 y = *(const float4*)(y1 + (long)i * 512 + 256 + c4);
  float4 b = *(const float4*)(b1 + c4);
  v4bf o;
  o[0] = (bf16_t)fmaxf(a.x * inv + y.x + b.x, 0.0f);
  o[1] = (bf16_t)fmaxf(a.y * inv + y.y + b.y, 0.0f);
  o[2] = (bf16_t)fmaxf(a.z * inv + y.z + b.z, 0.0f);
  o[3] = (bf16_t)fmaxf(a.w * inv + y.w + b.w, 0.0f);
  *(v4bf*)(zb + (long)i * 512 + 256 + c4) = o;
}

// zb[:,0:256] = bf16(agg/max(cnt,1))
__global__ void k_mean2(const float* __restrict__ agg, const float* __restrict__ cnt,
                        bf16_t* __restrict__ zb, int NN) {
  long t = (long)blockIdx.x * blockDim.x + threadIdx.x;
  long total = (long)NN * 64;
  if (t >= total) return;
  int i  = (int)(t >> 6);
  int c4 = (int)(t & 63) * 4;
  float inv = 1.0f / fmaxf(cnt[i], 1.0f);
  float4 a = *(const float4*)(agg + (long)i * 256 + c4);
  v4bf o;
  o[0] = (bf16_t)(a.x * inv);
  o[1] = (bf16_t)(a.y * inv);
  o[2] = (bf16_t)(a.z * inv);
  o[3] = (bf16_t)(a.w * inv);
  *(v4bf*)(zb + (long)i * 512 + c4) = o;
}

// ---------------- driver ----------------
extern "C" void kernel_launch(void* const* d_in, const int* in_sizes, int n_in,
                              void* d_out, int out_size, void* d_ws, size_t ws_size,
                              hipStream_t stream) {
  const float*     x   = (const float*)d_in[0];
  const long long* ei  = (const long long*)d_in[1];
  const float*     w1l = (const float*)d_in[2];
  const float*     w1r = (const float*)d_in[3];
  const float*     b1  = (const float*)d_in[4];
  const float*     w2l = (const float*)d_in[5];
  const float*     w2r = (const float*)d_in[6];
  const float*     b2  = (const float*)d_in[7];
  float* out = (float*)d_out;

  const int NN = in_sizes[0] / IN_DIM;   // 50000
  const int E  = in_sizes[1] / 2;        // 800000
  const long long* src = ei;
  const long long* dst = ei + E;

  char* ws = (char*)d_ws;
  auto take = [&](size_t bytes) { char* p = ws; ws += (bytes + 255) & ~(size_t)255; return p; };
  bf16_t* xb  = (bf16_t*)take((size_t)NN * IN_DIM * 2);   // x in bf16
  float*  y1  = (float*) take((size_t)NN * 512 * 4);      // [x@w1_l.T | x@w1_r.T]
  float*  agg = (float*) take((size_t)NN * 256 * 4);      // scatter accumulator (reused)
  bf16_t* zb  = (bf16_t*)take((size_t)NN * 512 * 2);      // [mean2 | h] in bf16
  float*  cnt = (float*) take((size_t)NN * 4);            // in-degree counts
  bf16_t* W1b = (bf16_t*)take((size_t)512 * 1024 * 2);    // [w1_l; w1_r] bf16
  bf16_t* W2b = (bf16_t*)take((size_t)512 * 512 * 2);     // [w2_l | w2_r] bf16

  // zero accumulators (workspace is poisoned, not zeroed, by the harness)
  k_zero_f32<<<2048, 256, 0, stream>>>(agg, (long)NN * 256);
  k_zero_f32<<<256, 256, 0, stream>>>(cnt, NN);

  // operand conversion / packing
  long n8x = (long)NN * IN_DIM / 8;
  k_cvt_bf16<<<(int)((n8x + 255) / 256), 256, 0, stream>>>(x, xb, n8x);
  long n8w1 = (long)512 * 1024 / 8;
  k_pack_rows_bf16<<<(int)((n8w1 + 255) / 256), 256, 0, stream>>>(w1l, w1r, 256, 1024, W1b);
  long n8w2 = (long)512 * 512 / 8;
  k_pack_cols_bf16<<<(int)((n8w2 + 255) / 256), 256, 0, stream>>>(w2l, w2r, 512, 256, W2b);

  // GEMM1: y1[NN,512] = xb @ W1b^T   (both linears of layer 1, pre-aggregation)
  dim3 g1(512 / 128, (NN + 127) / 128);
  k_gemm_bf16_wmma<<<g1, 256, 0, stream>>>(xb, W1b, y1, nullptr, NN, 512, IN_DIM);

  // layer-1 aggregation in projected (256-dim) space: lin_l(mean x) == mean(lin_l x)
  k_count_edges<<<(E + 255) / 256, 256, 0, stream>>>(dst, cnt, E);
  k_scatter_f32<<<1024, 256, 0, stream>>>(src, dst, y1, 512, 0, agg, E);

  // h = relu(mean + y1[:,256:] + b1) -> zb[:,256:]
  long nc = (long)NN * 64;
  k_combine1<<<(int)((nc + 255) / 256), 256, 0, stream>>>(y1, agg, cnt, b1, zb, NN);

  // layer-2 aggregation of h (bf16 gather halves the gather bytes)
  k_zero_f32<<<2048, 256, 0, stream>>>(agg, (long)NN * 256);
  k_scatter_bf16<<<1024, 256, 0, stream>>>(src, dst, zb, 512, 256, agg, E);
  k_mean2<<<(int)((nc + 255) / 256), 256, 0, stream>>>(agg, cnt, zb, NN);

  // GEMM2: out[NN,512] = zb @ W2b^T + b2   (K-concat fuses both linears of layer 2)
  k_gemm_bf16_wmma<<<g1, 256, 0, stream>>>(zb, W2b, out, b2, NN, 512, 512);
}
